// GAT_Conv_70257075028611
// MI455X (gfx1250) — matrix operand order repaired and checked by reading, not compiled, not run on hardware
//
#include <hip/hip_runtime.h>
#include <hip/hip_bf16.h>

// ---------------- WMMA vector types (gfx1250, wave32) ----------------
typedef __bf16 v16bf __attribute__((ext_vector_type(16)));
typedef __bf16 v8bf  __attribute__((ext_vector_type(8)));
typedef float  v8f   __attribute__((ext_vector_type(8)));

__device__ __forceinline__ unsigned short f32_to_bf16(float f) {
    unsigned u = __float_as_uint(f);
    u += 0x7FFFu + ((u >> 16) & 1u);          // round-to-nearest-even
    return (unsigned short)(u >> 16);
}

__device__ __forceinline__ float lrelu(float x) { return x > 0.f ? x : 0.2f * x; }

// Float atomic max via monotone int/uint mapping (init must be -inf).
__device__ __forceinline__ void atomicMaxF(float* a, float v) {
    if (v >= 0.f) atomicMax((int*)a, __float_as_int(v));
    else          atomicMin((unsigned int*)a, __float_as_uint(v));
}

// ---------------------------------------------------------------------
// GEMM: out[N,128] = x[N,fin] @ w[fin,128], bf16 WMMA with f32 accum.
// Block = 256 threads (8 waves) computes a 64x128 tile.
// Wave w: rows 16*(w>>1).., n-tiles 4*(w&1)..+3.
// ---------------------------------------------------------------------
__global__ void __launch_bounds__(256)
gemm_wmma_bf16(const float* __restrict__ x, const float* __restrict__ w,
               float* __restrict__ out, int nrows, int fin) {
    __shared__ unsigned short sW[128 * 128];   // W transposed: sW[n*fin + k]
    __shared__ unsigned short sX[64 * 128];    // X tile:       sX[r*fin + k]

    const int tid = threadIdx.x;
    const int block_row = blockIdx.x * 64;

    // Stage W (transposed) as bf16
    for (int idx = tid; idx < fin * 128; idx += 256) {
        int k = idx >> 7;          // idx / 128
        int n = idx & 127;
        sW[n * fin + k] = f32_to_bf16(w[k * 128 + n]);
    }
    // Stage X tile as bf16 (zero-pad past nrows)
    for (int idx = tid; idx < 64 * fin; idx += 256) {
        int r = idx / fin;
        int k = idx - r * fin;
        int gr = block_row + r;
        float v = (gr < nrows) ? x[(size_t)gr * fin + k] : 0.f;
        sX[r * fin + k] = f32_to_bf16(v);
    }
    __syncthreads();

    const int wave   = tid >> 5;
    const int lane   = tid & 31;
    const int half   = lane >> 4;        // 0: lanes 0-15, 1: lanes 16-31
    const int l16    = lane & 15;
    const int m_tile = wave >> 1;        // 0..3
    const int n_base = (wave & 1) * 4;   // 0 or 4

    v8f acc[4] = {};                     // four 16x16 f32 accumulators

    const int arow = m_tile * 16 + l16;  // A: lane holds row M = lane%16
    for (int kk = 0; kk < fin; kk += 32) {
        // A 16x32 bf16 layout: lane holds row M=lane%16, K=(half*16)..+15
        const v8bf a0 = *(const v8bf*)&sX[arow * fin + kk + half * 16];
        const v8bf a1 = *(const v8bf*)&sX[arow * fin + kk + half * 16 + 8];
        const v16bf a = __builtin_shufflevector(a0, a1,
            0,1,2,3,4,5,6,7,8,9,10,11,12,13,14,15);
        #pragma unroll
        for (int j = 0; j < 4; ++j) {
            // B 32x16 bf16 layout: lane holds col N=lane%16, K=(half*16)..+15
            const int col = (n_base + j) * 16 + l16;
            const v16bf b = *(const v16bf*)&sW[col * fin + kk + half * 16];
            acc[j] = __builtin_amdgcn_wmma_f32_16x16x32_bf16(
                false, a, false, b, (short)0, acc[j], false, false);
        }
    }

    // C/D layout: VGPR v -> M = v + 8*half, N = lane%16
    #pragma unroll
    for (int j = 0; j < 4; ++j) {
        const int col = (n_base + j) * 16 + l16;
        #pragma unroll
        for (int v = 0; v < 8; ++v) {
            const int grow = block_row + m_tile * 16 + half * 8 + v;
            if (grow < nrows) out[(size_t)grow * 128 + col] = acc[j][v];
        }
    }
}

// ---------------------------------------------------------------------
// Per-node attention halves: s[n,h] = <h[n,h,:], a_src[h,:]>, same for d.
// One wave per node; lane i covers cols 4i..4i+3; head = lane/16.
// ---------------------------------------------------------------------
__global__ void __launch_bounds__(256)
attn_coef(const float* __restrict__ h, const float* __restrict__ a_src,
          const float* __restrict__ a_dst, float* __restrict__ s,
          float* __restrict__ d, int n) {
    const int wave = (blockIdx.x * blockDim.x + threadIdx.x) >> 5;
    const int lane = threadIdx.x & 31;
    if (wave >= n) return;
    const float4 hv = *(const float4*)&h[(size_t)wave * 128 + lane * 4];
    const float4 as = *(const float4*)&a_src[lane * 4];  // [H,C] flat == col index
    const float4 ad = *(const float4*)&a_dst[lane * 4];
    float ps = hv.x*as.x + hv.y*as.y + hv.z*as.z + hv.w*as.w;
    float pd = hv.x*ad.x + hv.y*ad.y + hv.z*ad.z + hv.w*ad.w;
    #pragma unroll
    for (int m = 1; m <= 8; m <<= 1) {
        ps += __shfl_xor(ps, m, 32);
        pd += __shfl_xor(pd, m, 32);
    }
    if ((lane & 15) == 0) {
        const int hh = lane >> 4;
        s[wave * 2 + hh] = ps;
        d[wave * 2 + hh] = pd;
    }
}

__global__ void init_buffers(float* __restrict__ m, float* __restrict__ denom,
                             float* __restrict__ agg, int n) {
    const int i = blockIdx.x * blockDim.x + threadIdx.x;
    if (i < n * 2) { m[i] = -__builtin_inff(); denom[i] = 0.f; }
    if (i < n * 128) agg[i] = 0.f;
}

__device__ __forceinline__ void edge_nodes(const long long* ei, int e, int E,
                                           int& src, int& dst) {
    if (e < E) { src = (int)ei[e]; dst = (int)ei[(size_t)E + e]; }
    else       { src = dst = e - E; }   // self loop
}

__global__ void edge_max(const long long* __restrict__ ei,
                         const float* __restrict__ s, const float* __restrict__ d,
                         float* __restrict__ m, int E, int Etot) {
    const int e = blockIdx.x * blockDim.x + threadIdx.x;
    if (e >= Etot) return;
    int src, dst; edge_nodes(ei, e, E, src, dst);
    #pragma unroll
    for (int h = 0; h < 2; ++h) {
        const float v = lrelu(s[src * 2 + h] + d[dst * 2 + h]);
        atomicMaxF(&m[dst * 2 + h], v);
    }
}

__global__ void edge_expsum(const long long* __restrict__ ei,
                            const float* __restrict__ s, const float* __restrict__ d,
                            const float* __restrict__ m, float* __restrict__ denom,
                            int E, int Etot) {
    const int e = blockIdx.x * blockDim.x + threadIdx.x;
    if (e >= Etot) return;
    int src, dst; edge_nodes(ei, e, E, src, dst);
    #pragma unroll
    for (int h = 0; h < 2; ++h) {
        const float v = lrelu(s[src * 2 + h] + d[dst * 2 + h]);
        atomicAdd(&denom[dst * 2 + h], __expf(v - m[dst * 2 + h]));
    }
}

// One wave per edge; lane covers 4 features; head = lane/16.
__global__ void __launch_bounds__(256)
edge_aggregate(const long long* __restrict__ ei,
               const float* __restrict__ s, const float* __restrict__ d,
               const float* __restrict__ m, const float* __restrict__ denom,
               const float* __restrict__ hfeat, float* __restrict__ agg,
               int E, int Etot) {
    const int gw   = (blockIdx.x * blockDim.x + threadIdx.x) >> 5;
    const int lane = threadIdx.x & 31;
    if (gw >= Etot) return;
    int src, dst; edge_nodes(ei, gw, E, src, dst);
    const int hh = lane >> 4;
    const float v     = lrelu(s[src * 2 + hh] + d[dst * 2 + hh]);
    const float alpha = __expf(v - m[dst * 2 + hh]) / denom[dst * 2 + hh];
    const float4 hv = *(const float4*)&hfeat[(size_t)src * 128 + lane * 4];
    float* ap = &agg[(size_t)dst * 128 + lane * 4];
    atomicAdd(ap + 0, alpha * hv.x);
    atomicAdd(ap + 1, alpha * hv.y);
    atomicAdd(ap + 2, alpha * hv.z);
    atomicAdd(ap + 3, alpha * hv.w);
}

// Head mean + bias + ELU -> next layer input [N,64]
__global__ void epilogue(const float* __restrict__ agg, const float* __restrict__ b,
                         float* __restrict__ xout, int n) {
    const int i = blockIdx.x * blockDim.x + threadIdx.x;
    if (i >= n * 64) return;
    const int node = i >> 6, c = i & 63;
    const float v = 0.5f * (agg[(size_t)node * 128 + c] +
                            agg[(size_t)node * 128 + 64 + c]) + b[c];
    xout[i] = v > 0.f ? v : (__expf(v) - 1.f);   // ELU
}

// Final: sigmoid(x @ Wl + bl); one wave per node.
__global__ void __launch_bounds__(256)
final_linear(const float* __restrict__ x, const float* __restrict__ Wl,
             const float* __restrict__ bl, float* __restrict__ out, int n) {
    const int wave = (blockIdx.x * blockDim.x + threadIdx.x) >> 5;
    const int lane = threadIdx.x & 31;
    if (wave >= n) return;
    const float2 xv = *(const float2*)&x[(size_t)wave * 64 + lane * 2];
    const float2 wv = *(const float2*)&Wl[lane * 2];
    float p = xv.x * wv.x + xv.y * wv.y;
    #pragma unroll
    for (int mk = 1; mk < 32; mk <<= 1) p += __shfl_xor(p, mk, 32);
    if (lane == 0) out[wave] = 1.f / (1.f + __expf(-(p + bl[0])));
}

// ---------------------------------------------------------------------
extern "C" void kernel_launch(void* const* d_in, const int* in_sizes, int n_in,
                              void* d_out, int out_size, void* d_ws, size_t ws_size,
                              hipStream_t stream) {
    (void)n_in; (void)out_size; (void)ws_size;
    const float*     X  = (const float*)d_in[0];
    const long long* ei = (const long long*)d_in[1];   // int64 [2,E]
    // d_in[2] edge_weight: unused by reference

    const int N    = in_sizes[0] / 128;   // 50000
    const int E    = in_sizes[1] / 2;     // 800000
    const int Etot = E + N;               // + self loops

    float* ws    = (float*)d_ws;
    float* hfeat = ws;                                   // N*128
    float* agg   = hfeat + (size_t)N * 128;              // N*128
    float* xbuf  = agg   + (size_t)N * 128;              // N*64
    float* sat   = xbuf  + (size_t)N * 64;               // N*2
    float* dat   = sat   + (size_t)N * 2;                // N*2
    float* mmax  = dat   + (size_t)N * 2;                // N*2
    float* denom = mmax  + (size_t)N * 2;                // N*2
    float* out   = (float*)d_out;

    for (int layer = 0; layer < 3; ++layer) {
        const float* xin = (layer == 0) ? X : xbuf;
        const int    fin = (layer == 0) ? 128 : 64;
        const float* W  = (const float*)d_in[3 + 4 * layer];
        const float* as = (const float*)d_in[4 + 4 * layer];
        const float* ad = (const float*)d_in[5 + 4 * layer];
        const float* b  = (const float*)d_in[6 + 4 * layer];

        gemm_wmma_bf16<<<(N + 63) / 64, 256, 0, stream>>>(xin, W, hfeat, N, fin);
        attn_coef<<<(N + 7) / 8, 256, 0, stream>>>(hfeat, as, ad, sat, dat, N);
        init_buffers<<<((size_t)N * 128 + 255) / 256, 256, 0, stream>>>(mmax, denom, agg, N);
        edge_max<<<(Etot + 255) / 256, 256, 0, stream>>>(ei, sat, dat, mmax, E, Etot);
        edge_expsum<<<(Etot + 255) / 256, 256, 0, stream>>>(ei, sat, dat, mmax, denom, E, Etot);
        edge_aggregate<<<(Etot + 7) / 8, 256, 0, stream>>>(ei, sat, dat, mmax, denom,
                                                           hfeat, agg, E, Etot);
        epilogue<<<((size_t)N * 64 + 255) / 256, 256, 0, stream>>>(agg, b, xbuf, N);
    }
    final_linear<<<(N + 7) / 8, 256, 0, stream>>>(xbuf, (const float*)d_in[15],
                                                  (const float*)d_in[16], out, N);
}
